// DSPSModel_2276332667464
// MI455X (gfx1250) — matrix operand order repaired and checked by reading, not compiled, not run on hardware
//
#include <hip/hip_runtime.h>
#include <hip/hip_bf16.h>
#include <math.h>

// ---------------------------------------------------------------------------
// Model dims (match reference)
// ---------------------------------------------------------------------------
#define LSEQ 2048
#define DMODEL 1024
#define NHEAD 16
#define DHEAD 64
#define NLAYER 4
#define NCLS 64
#define NREL 32

typedef __attribute__((ext_vector_type(16))) __bf16 v16bf;
typedef __attribute__((ext_vector_type(8)))  float  v8f;

union ABu { v16bf v; unsigned int u[8]; };

static __device__ __forceinline__ unsigned short f2bf(float f) {
  unsigned int u = __float_as_uint(f);
  unsigned int r = u + 0x7FFFu + ((u >> 16) & 1u);   // round-to-nearest-even
  return (unsigned short)(r >> 16);
}
static __device__ __forceinline__ float bf2f(unsigned int us) {
  return __uint_as_float(us << 16);
}
static __device__ __forceinline__ float gelu_f(float x) {
  return 0.5f * x * (1.0f + erff(x * 0.70710678118654752f));
}
static __device__ __forceinline__ v8f zero8() {
  v8f z = {0.f,0.f,0.f,0.f,0.f,0.f,0.f,0.f};
  return z;
}
static __device__ __forceinline__ v8f wmma_bf16(v16bf a, v16bf b, v8f c) {
  return __builtin_amdgcn_wmma_f32_16x16x32_bf16(false, a, false, b, (short)0, c, false, false);
}

// ---- CDNA5 async global->LDS (ASYNCcnt path) ------------------------------
// LDS flat pointers carry the aperture tag in addr[63:32]; addr[31:0] is the
// LDS byte offset, so truncation yields the VDST operand.
static __device__ __forceinline__ void async16(unsigned short* lds, const unsigned short* g) {
  unsigned int la = (unsigned int)(unsigned long long)(uintptr_t)lds;
  unsigned long long ga = (unsigned long long)(uintptr_t)g;
  asm volatile("global_load_async_to_lds_b128 %0, %1, off" :: "v"(la), "v"(ga) : "memory");
}
static __device__ __forceinline__ void wait_async(void) {
  asm volatile("s_wait_asynccnt 0" ::: "memory");
}
static __device__ __forceinline__ void wait_async5(void) {
  asm volatile("s_wait_asynccnt 5" ::: "memory");
}

// A-fragment (16x32 bf16, M x K): lane r holds row r.
// VGPR v : K = 16*half + 2v,2v+1
static __device__ __forceinline__ v16bf load_a_frag(const unsigned short* base, int strideU) {
  int lane = threadIdx.x & 31, r = lane & 15, half = lane >> 4;
  const unsigned short* rowp = base + r * strideU;
  ABu u;
#pragma unroll
  for (int v = 0; v < 8; ++v) {
    int k = 16 * half + 2 * v;
    u.u[v] = *(const unsigned int*)(rowp + k);
  }
  return u.v;
}
// B-fragment for C = A * W^T : B[k][n] = W[n][k]; lane r holds column n=r.
static __device__ __forceinline__ v16bf load_b_fragT(const unsigned short* base, int strideU) {
  int lane = threadIdx.x & 31, r = lane & 15, half = lane >> 4;
  const unsigned short* rowp = base + r * strideU;
  ABu u;
#pragma unroll
  for (int v = 0; v < 8; ++v) {
    int k = 16 * half + 2 * v;
    u.u[v] = *(const unsigned int*)(rowp + k);
  }
  return u.v;
}

// ---------------------------------------------------------------------------
// WMMA GEMM: C[M,N] = act( A[M,K](bf16) @ W[N,K](bf16)^T + bias[N] )
// 256 thr (8 waves, 2x4) -> 64(M) x 256(N) block tile, wave tile 32x64,
// 32-K stages, double-buffered LDS fed by async global->LDS copies.
// ---------------------------------------------------------------------------
template <int EPI>   // 0 = none, 1 = exact GELU
__global__ __launch_bounds__(256) void k_gemm(
    const unsigned short* __restrict__ A, const unsigned short* __restrict__ W,
    const float* __restrict__ bias, float* __restrict__ C, int M, int N, int K) {
  __shared__ unsigned short Alds[2][64 * 40];
  __shared__ unsigned short Wlds[2][256 * 40];
  const int tid = threadIdx.x;
  const int wv = tid >> 5, lane = tid & 31, r = lane & 15, half = lane >> 4;
  const int m0 = blockIdx.y * 64, n0 = blockIdx.x * 256;
  const int mo = (wv >> 2) * 32, no = (wv & 3) * 64;

  // async-stage one 32-K slice into buffer b (5 async instructions per wave;
  // OOB rows are clamped, never skipped, so ASYNCcnt accounting stays exact)
  auto issue = [&](int b, int k0) {
    {
      int row = tid >> 2, ch = tid & 3;                 // A: 64 rows x 4 chunks
      int gm = m0 + row; if (gm > M - 1) gm = M - 1;
      async16(&Alds[b][row * 40 + ch * 8], A + (size_t)gm * K + k0 + ch * 8);
    }
#pragma unroll
    for (int i = 0; i < 4; ++i) {                       // W: 256 rows x 4 chunks
      int u = tid + i * 256;
      int row = u >> 2, ch = u & 3;
      int gn = n0 + row; if (gn > N - 1) gn = N - 1;
      async16(&Wlds[b][row * 40 + ch * 8], W + (size_t)gn * K + k0 + ch * 8);
    }
  };

  v8f acc[2][4];
#pragma unroll
  for (int mf = 0; mf < 2; ++mf)
#pragma unroll
    for (int nf = 0; nf < 4; ++nf) acc[mf][nf] = zero8();

  const int stages = K >> 5;
  issue(0, 0);
  for (int s = 0; s < stages; ++s) {
    __syncthreads();                      // everyone done reading buffer s-1
    if (s + 1 < stages) { issue((s + 1) & 1, (s + 1) << 5); wait_async5(); }
    else                { wait_async(); }
    __syncthreads();                      // stage-s tiles complete in all waves

    const unsigned short* Ab = &Alds[s & 1][0];
    const unsigned short* Wb = &Wlds[s & 1][0];
    v16bf a0 = load_a_frag(Ab + (mo + 0) * 40, 40);
    v16bf a1 = load_a_frag(Ab + (mo + 16) * 40, 40);
#pragma unroll
    for (int nf = 0; nf < 4; ++nf) {
      v16bf b = load_b_fragT(Wb + (no + nf * 16) * 40, 40);
      acc[0][nf] = wmma_bf16(a0, b, acc[0][nf]);
      acc[1][nf] = wmma_bf16(a1, b, acc[1][nf]);
    }
  }

#pragma unroll
  for (int mf = 0; mf < 2; ++mf)
#pragma unroll
    for (int nf = 0; nf < 4; ++nf) {
#pragma unroll
      for (int v = 0; v < 8; ++v) {
        int gm = m0 + mo + mf * 16 + v + 8 * half;
        int gn = n0 + no + nf * 16 + r;
        if (gm < M && gn < N) {
          float val = acc[mf][nf][v] + (bias ? bias[gn] : 0.0f);
          if (EPI == 1) val = gelu_f(val);
          C[(size_t)gm * N + gn] = val;
        }
      }
    }
}

// ---------------------------------------------------------------------------
// Fused flash attention, one layer: qkv bf16 [L,3D] -> o[L,D] (f32)
// grid (L/64, H), 128 thr (4 waves), each wave = 16 query rows, 32-key blocks.
// K/V/Q tiles stream in via async global->LDS.
// ---------------------------------------------------------------------------
__global__ __launch_bounds__(128) void k_attn(const unsigned short* __restrict__ qkv,
                                              float* __restrict__ o) {
  const int h = blockIdx.y;
  const int qb0 = blockIdx.x * 64;
  const int tid = threadIdx.x;
  const int wv = tid >> 5, lane = tid & 31, r = lane & 15, half = lane >> 4;
  const float scale = 0.125f;   // 1/sqrt(64)

  __shared__ unsigned short Qs[64 * 72];
  __shared__ unsigned short Ks[32 * 72];
  __shared__ unsigned short Vs[32 * 72];
  __shared__ unsigned short Ps[4][16 * 40];

  // Q tile: 64 rows x 8 16B-chunks, async
#pragma unroll
  for (int i = 0; i < 4; ++i) {
    int u = tid + i * 128;
    int row = u >> 3, ch = u & 7;
    async16(&Qs[row * 72 + ch * 8],
            qkv + (size_t)(qb0 + row) * 3072 + h * 64 + ch * 8);
  }

  v8f O0 = zero8(), O1 = zero8(), O2 = zero8(), O3 = zero8();
  float m_[8], l_[8];
#pragma unroll
  for (int v = 0; v < 8; ++v) { m_[v] = -3.0e38f; l_[v] = 0.0f; }

  for (int kb = 0; kb < LSEQ / 32; ++kb) {
    __syncthreads();
#pragma unroll
    for (int i = 0; i < 2; ++i) {         // K tile: 32 rows x 8 chunks
      int u = tid + i * 128;
      int row = u >> 3, ch = u & 7;
      size_t key = (size_t)(kb * 32 + row) * 3072;
      async16(&Ks[row * 72 + ch * 8], qkv + key + 1024 + h * 64 + ch * 8);
    }
#pragma unroll
    for (int i = 0; i < 2; ++i) {         // V tile
      int u = tid + i * 128;
      int row = u >> 3, ch = u & 7;
      size_t key = (size_t)(kb * 32 + row) * 3072;
      async16(&Vs[row * 72 + ch * 8], qkv + key + 2048 + h * 64 + ch * 8);
    }
    wait_async();                          // also covers Q on first iteration
    __syncthreads();

    // S = Q K^T  (16x32 per wave, K-dim = 64 -> two wmmas per n-frag)
    v8f s0 = zero8(), s1 = zero8();
    const unsigned short* Qbase = Qs + (wv * 16) * 72;
#pragma unroll
    for (int ko = 0; ko < 64; ko += 32) {
      v16bf a  = load_a_frag(Qbase + ko, 72);
      v16bf b0 = load_b_fragT(Ks + 0 * 72 + ko, 72);
      v16bf b1 = load_b_fragT(Ks + 16 * 72 + ko, 72);
      s0 = wmma_bf16(a, b0, s0);
      s1 = wmma_bf16(a, b1, s1);
    }

    // online softmax over the 32 keys, stats per row (v, half)
#pragma unroll
    for (int v = 0; v < 8; ++v) {
      float a0 = s0[v] * scale, a1 = s1[v] * scale;
      float mx = fmaxf(a0, a1);
#pragma unroll
      for (int msk = 8; msk >= 1; msk >>= 1) mx = fmaxf(mx, __shfl_xor(mx, msk, 32));
      float mn = fmaxf(m_[v], mx);
      float alpha = expf(m_[v] - mn);
      float p0 = expf(a0 - mn), p1 = expf(a1 - mn);
      float rs = p0 + p1;
#pragma unroll
      for (int msk = 8; msk >= 1; msk >>= 1) rs += __shfl_xor(rs, msk, 32);
      l_[v] = l_[v] * alpha + rs;
      m_[v] = mn;
      O0[v] *= alpha; O1[v] *= alpha; O2[v] *= alpha; O3[v] *= alpha;
      int prow = v + 8 * half;
      Ps[wv][prow * 40 + r]      = f2bf(p0);   // N = 0..15
      Ps[wv][prow * 40 + 16 + r] = f2bf(p1);   // N = 16..31
    }
    __syncthreads();   // LDS ordering for P restage (C-layout -> A-layout)

    // O += P(16x32) @ V(32x64)
    v16bf pa = load_a_frag(Ps[wv], 40);
    v8f* Oc[4] = {&O0, &O1, &O2, &O3};
#pragma unroll
    for (int c = 0; c < 4; ++c) {
      ABu u;
#pragma unroll
      for (int v = 0; v < 8; ++v) {
        int k = 16 * half + 2 * v;
        unsigned int lo = Vs[k * 72 + c * 16 + r];
        unsigned int hi = Vs[(k + 1) * 72 + c * 16 + r];
        u.u[v] = lo | (hi << 16);
      }
      *Oc[c] = wmma_bf16(pa, u.v, *Oc[c]);
    }
  }

  v8f* Oc[4] = {&O0, &O1, &O2, &O3};
#pragma unroll
  for (int c = 0; c < 4; ++c)
#pragma unroll
    for (int v = 0; v < 8; ++v) {
      int mrow = qb0 + wv * 16 + v + 8 * half;
      o[(size_t)mrow * DMODEL + h * 64 + c * 16 + r] = (*Oc[c])[v] / l_[v];
    }
}

// ---------------------------------------------------------------------------
// LayerNorm (D=1024, 256 thr/row): out = LN(x (+res)) * g + b
// ---------------------------------------------------------------------------
__global__ __launch_bounds__(256) void k_ln(const float* __restrict__ x,
    const float* __restrict__ res, const float* __restrict__ g,
    const float* __restrict__ b, float* __restrict__ out) {
  int row = blockIdx.x;
  __shared__ float red[8];
  const float* xr = x + (size_t)row * DMODEL;
  const float* rr = res ? res + (size_t)row * DMODEL : nullptr;
  float v[4]; float s = 0.f;
#pragma unroll
  for (int i = 0; i < 4; ++i) {
    int c = threadIdx.x + i * 256;
    float t = xr[c]; if (rr) t += rr[c];
    v[i] = t; s += t;
  }
#pragma unroll
  for (int m = 16; m >= 1; m >>= 1) s += __shfl_xor(s, m, 32);
  if ((threadIdx.x & 31) == 0) red[threadIdx.x >> 5] = s;
  __syncthreads();
  float mean = 0.f;
#pragma unroll
  for (int i = 0; i < 8; ++i) mean += red[i];
  mean *= (1.0f / DMODEL);
  __syncthreads();
  float q = 0.f;
#pragma unroll
  for (int i = 0; i < 4; ++i) { float d = v[i] - mean; q += d * d; }
#pragma unroll
  for (int m = 16; m >= 1; m >>= 1) q += __shfl_xor(q, m, 32);
  if ((threadIdx.x & 31) == 0) red[threadIdx.x >> 5] = q;
  __syncthreads();
  float var = 0.f;
#pragma unroll
  for (int i = 0; i < 8; ++i) var += red[i];
  var *= (1.0f / DMODEL);
  float inv = rsqrtf(var + 1e-5f);
#pragma unroll
  for (int i = 0; i < 4; ++i) {
    int c = threadIdx.x + i * 256;
    out[(size_t)row * DMODEL + c] = (v[i] - mean) * inv * g[c] + b[c];
  }
}

// f32 -> bf16 elementwise
__global__ void k_f2bf(const float* __restrict__ src, unsigned short* __restrict__ dst, int n) {
  int i = blockIdx.x * 256 + threadIdx.x;
  if (i < n) dst[i] = f2bf(src[i]);
}

// row softmax, 64 cols, one wave per row
__global__ __launch_bounds__(32) void k_softmax64(const float* __restrict__ in,
                                                  float* __restrict__ out) {
  int row = blockIdx.x, t = threadIdx.x;
  float a = in[(size_t)row * 64 + t], b = in[(size_t)row * 64 + 32 + t];
  float mx = fmaxf(a, b);
#pragma unroll
  for (int m = 16; m >= 1; m >>= 1) mx = fmaxf(mx, __shfl_xor(mx, m, 32));
  float ea = expf(a - mx), eb = expf(b - mx);
  float s = ea + eb;
#pragma unroll
  for (int m = 16; m >= 1; m >>= 1) s += __shfl_xor(s, m, 32);
  out[(size_t)row * 64 + t] = ea / s;
  out[(size_t)row * 64 + 32 + t] = eb / s;
}

// column mean over L rows -> root[D]
__global__ void k_colmean(const float* __restrict__ x, float* __restrict__ root) {
  int c = blockIdx.x * 256 + threadIdx.x;
  if (c < DMODEL) {
    float s = 0.f;
    for (int i = 0; i < LSEQ; ++i) s += x[(size_t)i * DMODEL + c];
    root[c] = s * (1.0f / LSEQ);
  }
}

// GRU scan: one WG, thread t owns h[t]; Whh in bf16 (L2-resident)
__global__ __launch_bounds__(1024) void k_gru(const float* __restrict__ Gi,
    const unsigned short* __restrict__ Whh, const float* __restrict__ bhh,
    float* __restrict__ hseq) {
  __shared__ float hsh[DMODEL];
  int t = threadIdx.x;
  hsh[t] = 0.0f;
  __syncthreads();
  const unsigned short* r0 = Whh + (size_t)t * DMODEL;
  const unsigned short* r1 = Whh + (size_t)(DMODEL + t) * DMODEL;
  const unsigned short* r2 = Whh + (size_t)(2 * DMODEL + t) * DMODEL;
  float b0 = bhh[t], b1 = bhh[DMODEL + t], b2 = bhh[2 * DMODEL + t];
  for (int step = 0; step < LSEQ; ++step) {
    float hr = b0, hz = b1, hn = b2;
    for (int j = 0; j < DMODEL; j += 2) {
      float h0 = hsh[j], h1 = hsh[j + 1];
      unsigned int w0 = *(const unsigned int*)(r0 + j);
      unsigned int w1 = *(const unsigned int*)(r1 + j);
      unsigned int w2 = *(const unsigned int*)(r2 + j);
      hr += bf2f(w0 & 0xffffu) * h0 + bf2f(w0 >> 16) * h1;
      hz += bf2f(w1 & 0xffffu) * h0 + bf2f(w1 >> 16) * h1;
      hn += bf2f(w2 & 0xffffu) * h0 + bf2f(w2 >> 16) * h1;
    }
    const float* gi = Gi + (size_t)step * 3 * DMODEL;
    float rg = 1.f / (1.f + expf(-(gi[t] + hr)));
    float zg = 1.f / (1.f + expf(-(gi[DMODEL + t] + hz)));
    float ng = tanhf(gi[2 * DMODEL + t] + rg * hn);
    float hprev = hsh[t];
    float hnew = (1.f - zg) * ng + zg * hprev;
    __syncthreads();
    hsh[t] = hnew;
    hseq[(size_t)step * DMODEL + t] = hnew;
    __syncthreads();
  }
}

// rootplus bf16: row 0 = root, rows 1..L = hseq
__global__ void k_rootplus(const float* __restrict__ root, const float* __restrict__ hseq,
                           unsigned short* __restrict__ out) {
  int j = blockIdx.x;
  const float* src = (j == 0) ? root : hseq + (size_t)(j - 1) * DMODEL;
  for (int c = threadIdx.x; c < DMODEL; c += 256)
    out[(size_t)j * DMODEL + c] = f2bf(src[c]);
}

// Atab[j] = ppar[j] @ M_cp   (j=0 -> row C of M_cp; j>=1 -> cls_prob[j-1] @ M_cp[0:C])
__global__ __launch_bounds__(64) void k_partab(const float* __restrict__ cp,
    const float* __restrict__ Mcp, float* __restrict__ Atab) {
  int j = blockIdx.x, c = threadIdx.x;
  float s;
  if (j == 0) s = Mcp[NCLS * NCLS + c];
  else {
    s = 0.f;
    for (int k = 0; k < NCLS; ++k) s += cp[(size_t)(j - 1) * NCLS + k] * Mcp[k * NCLS + c];
  }
  Atab[(size_t)j * NCLS + c] = s;
}

// par_logits[i][j] = (j<=i) ? score[i][j] + log(Atab[j].cp[i] + eps) : -1e9
__global__ __launch_bounds__(256) void k_parlogits(const float* __restrict__ score,
    const float* __restrict__ Atab, const float* __restrict__ cp, float* __restrict__ out) {
  int i = blockIdx.x;
  __shared__ float ci[NCLS];
  if (threadIdx.x < NCLS) ci[threadIdx.x] = cp[(size_t)i * NCLS + threadIdx.x];
  __syncthreads();
  for (int j = threadIdx.x; j <= LSEQ; j += 256) {
    float val;
    if (j <= i) {
      float d = 0.f;
      for (int c = 0; c < NCLS; ++c) d += Atab[(size_t)j * NCLS + c] * ci[c];
      val = score[(size_t)i * (LSEQ + 1) + j] + logf(d + 1e-8f);
    } else val = -1e9f;
    out[(size_t)i * (LSEQ + 1) + j] = val;
  }
}

// feat bf16 [L, 2D] = [h_seq | parent-vec]
__global__ void k_feat(const float* __restrict__ hseq, const float* __restrict__ root,
                       const int* __restrict__ yp, unsigned short* __restrict__ feat) {
  int i = blockIdx.x;
  int p = yp[i];
  int pc = p < 0 ? 0 : (p > LSEQ - 1 ? LSEQ - 1 : p);
  const float* pv = (p < 0) ? root : hseq + (size_t)pc * DMODEL;
  for (int c = threadIdx.x; c < 2 * DMODEL; c += 256) {
    float v = (c < DMODEL) ? hseq[(size_t)i * DMODEL + c] : pv[c - DMODEL];
    feat[(size_t)i * 2 * DMODEL + c] = f2bf(v);
  }
}

// ---------------------------------------------------------------------------
// Launch
// ---------------------------------------------------------------------------
extern "C" void kernel_launch(void* const* d_in, const int* in_sizes, int n_in,
                              void* d_out, int out_size, void* d_ws, size_t ws_size,
                              hipStream_t stream) {
  (void)in_sizes; (void)n_in; (void)out_size; (void)ws_size;
  const int L = LSEQ, D = DMODEL, C = NCLS, R = NREL;

  const float* x       = (const float*)d_in[0];
  const float* fuse_g  = (const float*)d_in[1];
  const float* fuse_b  = (const float*)d_in[2];
  const float* eWqkv   = (const float*)d_in[3];
  const float* ebqkv   = (const float*)d_in[4];
  const float* eWo     = (const float*)d_in[5];
  const float* ebo     = (const float*)d_in[6];
  const float* eln1g   = (const float*)d_in[7];
  const float* eln1b   = (const float*)d_in[8];
  const float* eW1     = (const float*)d_in[9];
  const float* eb1     = (const float*)d_in[10];
  const float* eW2     = (const float*)d_in[11];
  const float* eb2     = (const float*)d_in[12];
  const float* eln2g   = (const float*)d_in[13];
  const float* eln2b   = (const float*)d_in[14];
  const float* cls_W   = (const float*)d_in[15];
  const float* cls_b   = (const float*)d_in[16];
  const float* gWih    = (const float*)d_in[17];
  const float* gWhh    = (const float*)d_in[18];
  const float* gbih    = (const float*)d_in[19];
  const float* gbhh    = (const float*)d_in[20];
  const float* Wq      = (const float*)d_in[21];
  const float* Wk      = (const float*)d_in[22];
  const float* rW1     = (const float*)d_in[23];
  const float* rb1     = (const float*)d_in[24];
  const float* rW2     = (const float*)d_in[25];
  const float* rb2     = (const float*)d_in[26];
  const float* Mcp     = (const float*)d_in[27];
  const int*   ypar    = (const int*)d_in[28];

  float* out = (float*)d_out;
  float* out_cls = out;                                   // [L,C]
  float* out_par = out + (size_t)L * C;                   // [L,L+1]
  float* out_rel = out_par + (size_t)L * (L + 1);         // [L,R]

  // --- workspace bump allocator (~176 MB total) ---
  char* base = (char*)d_ws; size_t cur = 0;
  auto take = [&](size_t bytes) -> void* {
    void* p = base + cur; cur = (cur + bytes + 255) & ~(size_t)255; return p;
  };
  float* h     = (float*)take((size_t)L * D * 4);
  float* qkvb  = (float*)take((size_t)L * 3 * D * 4);     // also reused as proj / r1
  float* attno = (float*)take((size_t)L * D * 4);
  float* tmp4  = (float*)take((size_t)L * 4 * D * 4);
  unsigned short* abf  = (unsigned short*)take((size_t)L * 4 * D * 2); // act bf16 staging
  unsigned short* wbf  = (unsigned short*)take((size_t)4 * D * D * 2); // weight bf16 staging
  float* hseq  = (float*)take((size_t)L * D * 4);
  float* Gi    = (float*)take((size_t)L * 3 * D * 4);
  float* cprob = (float*)take((size_t)L * C * 4);
  float* Atab  = (float*)take((size_t)(L + 1) * C * 4);
  float* qb    = (float*)take((size_t)L * D * 4);
  float* keys  = (float*)take((size_t)(L + 1) * D * 4);
  float* score = (float*)take((size_t)L * (L + 1) * 4);
  float* root  = (float*)take((size_t)D * 4);
  unsigned short* whhb = (unsigned short*)take((size_t)3 * D * D * 2);
  float* proj = qkvb;   // reuse

  auto cvt = [&](const float* s, unsigned short* d, size_t n) {
    k_f2bf<<<(int)((n + 255) / 256), 256, 0, stream>>>(s, d, (int)n);
  };
  auto gemm = [&](const unsigned short* A, const unsigned short* W, const float* bias,
                  float* Cp, int M, int N, int K, int epi) {
    dim3 g((N + 255) / 256, (M + 63) / 64);
    if (epi) k_gemm<1><<<g, 256, 0, stream>>>(A, W, bias, Cp, M, N, K);
    else     k_gemm<0><<<g, 256, 0, stream>>>(A, W, bias, Cp, M, N, K);
  };

  // fuse LayerNorm
  k_ln<<<L, 256, 0, stream>>>(x, nullptr, fuse_g, fuse_b, h);

  // encoder layers
  for (int l = 0; l < NLAYER; ++l) {
    cvt(h, abf, (size_t)L * D);
    cvt(eWqkv + (size_t)l * 3 * D * D, wbf, (size_t)3 * D * D);
    gemm(abf, wbf, ebqkv + (size_t)l * 3 * D, qkvb, L, 3 * D, D, 0);
    cvt(qkvb, abf, (size_t)L * 3 * D);                    // qkv -> bf16 for attention
    k_attn<<<dim3(L / 64, NHEAD), 128, 0, stream>>>(abf, attno);
    cvt(attno, abf, (size_t)L * D);
    cvt(eWo + (size_t)l * D * D, wbf, (size_t)D * D);
    gemm(abf, wbf, ebo + (size_t)l * D, proj, L, D, D, 0);
    k_ln<<<L, 256, 0, stream>>>(h, proj, eln1g + (size_t)l * D, eln1b + (size_t)l * D, h);
    cvt(h, abf, (size_t)L * D);
    cvt(eW1 + (size_t)l * 4 * D * D, wbf, (size_t)4 * D * D);
    gemm(abf, wbf, eb1 + (size_t)l * 4 * D, tmp4, L, 4 * D, D, 1);   // GELU epilogue
    cvt(tmp4, abf, (size_t)L * 4 * D);
    cvt(eW2 + (size_t)l * D * 4 * D, wbf, (size_t)4 * D * D);
    gemm(abf, wbf, eb2 + (size_t)l * D, proj, L, D, 4 * D, 0);
    k_ln<<<L, 256, 0, stream>>>(h, proj, eln2g + (size_t)l * D, eln2b + (size_t)l * D, h);
  }

  // classifier + softmax + root mean
  cvt(h, abf, (size_t)L * D);
  cvt(cls_W, wbf, (size_t)C * D);
  gemm(abf, wbf, cls_b, out_cls, L, C, D, 0);
  k_softmax64<<<L, 32, 0, stream>>>(out_cls, cprob);
  k_colmean<<<(D + 255) / 256, 256, 0, stream>>>(h, root);

  // GRU: Gi = x_star @ Wih^T + bih, then sequential scan
  cvt(gWih, wbf, (size_t)3 * D * D);
  gemm(abf, wbf, gbih, Gi, L, 3 * D, D, 0);
  cvt(gWhh, whhb, (size_t)3 * D * D);
  k_gru<<<1, 1024, 0, stream>>>(Gi, whhb, gbhh, hseq);

  // q / keys / score
  k_rootplus<<<L + 1, 256, 0, stream>>>(root, hseq, abf);   // abf = [root; hseq] bf16
  cvt(Wq, wbf, (size_t)D * D);
  gemm(abf + D, wbf, nullptr, qb, L, D, D, 0);              // rows 1..L = hseq
  cvt(Wk, wbf, (size_t)D * D);
  gemm(abf, wbf, nullptr, keys, L + 1, D, D, 0);
  cvt(qb, abf, (size_t)L * D);
  cvt(keys, wbf, (size_t)(L + 1) * D);
  gemm(abf, wbf, nullptr, score, L, L + 1, D, 0);           // score = q @ keys^T

  // parent logits with prior + causal mask
  k_partab<<<L + 1, 64, 0, stream>>>(cprob, Mcp, Atab);
  k_parlogits<<<L, 256, 0, stream>>>(score, Atab, cprob, out_par);

  // relation head
  k_feat<<<L, 256, 0, stream>>>(hseq, root, ypar, abf);     // feat bf16 [L,2D]
  cvt(rW1, wbf, (size_t)D * 2 * D);
  gemm(abf, wbf, rb1, proj, L, D, 2 * D, 1);                // GELU epilogue -> r1
  cvt(proj, abf, (size_t)L * D);
  cvt(rW2, wbf, (size_t)R * D);
  gemm(abf, wbf, rb2, out_rel, L, R, D, 0);
}
